// MaskedMultiHeadAttention_91070486544806
// MI455X (gfx1250) — matrix operand order repaired and checked by reading, not compiled, not run on hardware
//
#include <hip/hip_runtime.h>

#define DIM   2048
#define NH    16
#define HD    128
#define BATCH 4
#define SEQ   2048
#define ROWS  (BATCH * SEQ)

typedef __bf16 bf16;
typedef __attribute__((ext_vector_type(16))) __bf16 v16bf;
typedef __attribute__((ext_vector_type(8)))  float  v8f;
typedef __attribute__((ext_vector_type(4))) unsigned int u32x4;
typedef __attribute__((ext_vector_type(8))) unsigned int u32x8;

static __device__ inline v8f wmma_bf16(v16bf a, v16bf b, v8f c) {
  // D = A(16x32 bf16) * B(32x16 bf16) + C(16x16 f32)
  return __builtin_amdgcn_wmma_f32_16x16x32_bf16(false, a, false, b, (short)0, c,
                                                 false, false);
}

static __device__ inline v8f vzero8() {
  v8f z;
#pragma unroll
  for (int i = 0; i < 8; ++i) z[i] = 0.0f;
  return z;
}

// A fragment: 16x32 (MxK) bf16, row-major source with leading dim ld (elements).
// lane holds row M=lane&15; K chunks {kb..kb+7, kb+16..kb+23}, kb = 8*(lane>>4).
static __device__ inline v16bf load_fragA(const bf16* base, int ld) {
  const int lane = threadIdx.x & 31;
  const bf16* p = base + (lane & 15) * ld + ((lane >> 4) << 3);
  union { v16bf v; uint4 q[2]; } f;
  f.q[0] = *(const uint4*)(p);
  f.q[1] = *(const uint4*)(p + 16);
  return f.v;
}

// B fragment: 32x16 (KxN) bf16 where memory holds B^T row-major (row n, contiguous k).
// lane holds col N=lane&15; K chunks {kb..kb+7, kb+16..kb+23}, kb = 8*(lane>>4).
static __device__ inline v16bf load_fragB(const bf16* base, int ld) {
  const int lane = threadIdx.x & 31;
  const bf16* p = base + (lane & 15) * ld + ((lane >> 4) << 3);
  union { v16bf v; uint4 q[2]; } f;
  f.q[0] = *(const uint4*)(p);
  f.q[1] = *(const uint4*)(p + 16);
  return f.v;
}

// Tensor Data Mover: async-DMA a 2D bf16 tile (tile_d0 contiguous elems x tile_d1
// rows, row stride stride0 elems) from global memory into LDS at lds_off.
// D# packed per CDNA5 ISA 8.3/8.4. pad: +16B after every 64B stored -> LDS row
// stride of 40 bf16 elements (matches load_fragA/B above). Issued by one wave,
// tracked with TENSORcnt.
static __device__ inline void tdm_load_2d(unsigned lds_off, const void* gaddr,
                                          unsigned tile_d0, unsigned tile_d1,
                                          unsigned tensor_d0, unsigned tensor_d1,
                                          unsigned stride0) {
  unsigned long long ga = (unsigned long long)(uintptr_t)gaddr;
  u32x4 g0;
  g0[0] = 1u;                                   // count=1, user descriptor
  g0[1] = lds_off;                              // lds_addr (bytes)
  g0[2] = (unsigned)ga;                         // global_addr[31:0]
  g0[3] = (unsigned)((ga >> 32) & 0x01FFFFFFu)  // global_addr[56:32]
          | 0x80000000u;                        // type=2 ("image")
  u32x8 g1;
  g1[0] = (1u << 16)                            // data_size = 2 bytes
        | (1u << 20)                            // pad_enable
        | (3u << 22)                            // pad_interval: 16 DWORDs (64B)
        | (3u << 25);                           // pad_amount: 4 DWORDs (16B)
  g1[1] = (tensor_d0 & 0xFFFFu) << 16;          // tensor_dim0[15:0]
  g1[2] = (tensor_d0 >> 16) | ((tensor_d1 & 0xFFFFu) << 16);
  g1[3] = (tensor_d1 >> 16) | (tile_d0 << 16);  // tile_dim0
  g1[4] = tile_d1;                              // tile_dim1 (tile_dim2 = 0)
  g1[5] = stride0;                              // tensor_dim0_stride[31:0]
  g1[6] = 0u;
  g1[7] = 0u;
  asm volatile("tensor_load_to_lds %0, %1" : : "s"(g0), "s"(g1) : "memory");
}

__global__ __launch_bounds__(256) void cast_f32_bf16(const float* __restrict__ in,
                                                     bf16* __restrict__ out, int n4) {
  int i = blockIdx.x * blockDim.x + threadIdx.x;
  if (i >= n4) return;
  float4 f = ((const float4*)in)[i];
  union { bf16 b[4]; uint2 u; } o;
  o.b[0] = (bf16)f.x; o.b[1] = (bf16)f.y; o.b[2] = (bf16)f.z; o.b[3] = (bf16)f.w;
  ((uint2*)out)[i] = o.u;
}

// C = A(row-major bf16, ROWSxDIM) * W^T (W row-major bf16, DIMxDIM), 64x128 tile.
// Staging: TDM tensor_load_to_lds, double-buffered, wave 0 issues, TENSORcnt sync.
// mode 0: Q -> RoPE -> [B,H,S,Hd] bf16      mode 1: K -> RoPE -> [B,H,S,Hd] bf16
// mode 2: V -> [B,H,Hd,S] bf16 (transposed) mode 3: out = C + bias -> f32 [B,S,D]
__global__ __launch_bounds__(256)
void gemm_qkvo(const bf16* __restrict__ A,
               const bf16* __restrict__ W0, const bf16* __restrict__ W1,
               const bf16* __restrict__ W2, const bf16* __restrict__ W3,
               bf16* __restrict__ Qo, bf16* __restrict__ Ko, bf16* __restrict__ Vto,
               float* __restrict__ Fo, const float* __restrict__ bias, int mode0) {
  __shared__ bf16  Xs[2][64 * 40];
  __shared__ bf16  Ws[2][128 * 40];
  __shared__ float Cs[64 * 128];

  const int mode = mode0 + (int)blockIdx.z;
  const bf16* W = (mode == 0) ? W0 : (mode == 1) ? W1 : (mode == 2) ? W2 : W3;
  const int m0 = blockIdx.x * 64;
  const int n0 = blockIdx.y * 128;
  const int t = threadIdx.x;
  const int wave = t >> 5, lane = t & 31;
  const int lhalf = lane >> 4, cn = lane & 15;
  const int wm = wave >> 2;  // 0..1 : M sub-tile (32 rows)
  const int wn = wave & 3;   // 0..3 : N sub-tile (32 cols)

  const unsigned xo0 = (unsigned)(uintptr_t)&Xs[0][0];
  const unsigned xo1 = (unsigned)(uintptr_t)&Xs[1][0];
  const unsigned wo0 = (unsigned)(uintptr_t)&Ws[0][0];
  const unsigned wo1 = (unsigned)(uintptr_t)&Ws[1][0];
  const bf16* Ab = A + (size_t)m0 * DIM;
  const bf16* Wb = W + (size_t)n0 * DIM;

  v8f acc[2][2];
  acc[0][0] = vzero8(); acc[0][1] = vzero8();
  acc[1][0] = vzero8(); acc[1][1] = vzero8();

  if (wave == 0) {  // prologue: stage k-tile 0 into buffer 0
    tdm_load_2d(xo0, Ab, 32, 64, DIM, (unsigned)(ROWS - m0), DIM);
    tdm_load_2d(wo0, Wb, 32, 128, DIM, (unsigned)(DIM - n0), DIM);
  }

  int buf = 0;
  for (int kk = 0; kk < DIM; kk += 32, buf ^= 1) {
    if (wave == 0) {
      if (kk + 32 < DIM) {  // stage next tile into the other buffer
        tdm_load_2d(buf ? xo0 : xo1, Ab + kk + 32, 32, 64,
                    (unsigned)(DIM - kk - 32), (unsigned)(ROWS - m0), DIM);
        tdm_load_2d(buf ? wo0 : wo1, Wb + kk + 32, 32, 128,
                    (unsigned)(DIM - kk - 32), (unsigned)(DIM - n0), DIM);
        __builtin_amdgcn_s_wait_tensorcnt(2);  // current tile's 2 DMAs done
      } else {
        __builtin_amdgcn_s_wait_tensorcnt(0);
      }
    }
    __syncthreads();

    const bf16* Xb_ = buf ? &Xs[1][0] : &Xs[0][0];
    const bf16* Wb_ = buf ? &Ws[1][0] : &Ws[0][0];
    v16bf a0 = load_fragA(Xb_ + (wm * 32) * 40, 40);
    v16bf a1 = load_fragA(Xb_ + (wm * 32 + 16) * 40, 40);
    v16bf b0 = load_fragB(Wb_ + (wn * 32) * 40, 40);
    v16bf b1 = load_fragB(Wb_ + (wn * 32 + 16) * 40, 40);
    acc[0][0] = wmma_bf16(a0, b0, acc[0][0]);
    acc[0][1] = wmma_bf16(a0, b1, acc[0][1]);
    acc[1][0] = wmma_bf16(a1, b0, acc[1][0]);
    acc[1][1] = wmma_bf16(a1, b1, acc[1][1]);
    __syncthreads();  // all waves done reading 'buf' before it is re-filled
  }

  // Stage accumulators to LDS (C layout: row = v + 8*half, col = lane&15).
#pragma unroll
  for (int mi = 0; mi < 2; ++mi)
#pragma unroll
    for (int ni = 0; ni < 2; ++ni)
#pragma unroll
      for (int v = 0; v < 8; ++v)
        Cs[(wm * 32 + mi * 16 + lhalf * 8 + v) * 128 + (wn * 32 + ni * 16 + cn)] =
            acc[mi][ni][v];
  __syncthreads();

  const int h = blockIdx.y;  // N tile == one head for modes 0..2
#pragma unroll 4
  for (int i = 0; i < 32; ++i) {
    int e = t + i * 256;       // 64*128 elements / 256 threads
    int m = e >> 7, n = e & 127;
    int gm = m0 + m;
    int b = gm >> 11, s = gm & (SEQ - 1);
    float val = Cs[m * 128 + n];
    if (mode <= 1) {  // RoPE(d pairs with d+-64), head-local dim n
      int d2 = (n < 64) ? n : n - 64;
      float invf = __expf(-((float)(2 * d2) * (1.0f / 128.0f)) * 9.210340371976184f);
      float ang = (float)s * invf;
      float c = __cosf(ang), sn = __sinf(ang);
      float other = (n < 64) ? Cs[m * 128 + n + 64] : Cs[m * 128 + n - 64];
      val = (n < 64) ? (val * c - other * sn) : (val * c + other * sn);
      bf16* dst = (mode == 0) ? Qo : Ko;
      dst[((size_t)(b * NH + h) * SEQ + s) * HD + n] = (bf16)val;
    } else if (mode == 2) {
      Vto[((size_t)(b * NH + h) * HD + n) * SEQ + s] = (bf16)val;
    } else {
      Fo[(size_t)gm * DIM + n0 + n] = val + bias[n0 + n];
    }
  }
}

// Flash attention, causal. One wave = 16 q rows, fully independent (no block barriers).
__global__ __launch_bounds__(256)
void flash_attn(const bf16* __restrict__ Q, const bf16* __restrict__ K,
                const bf16* __restrict__ Vt, bf16* __restrict__ O) {
  __shared__ bf16 Ps[8 * 16 * 32];  // per-wave P transpose staging
  const int t = threadIdx.x, wave = t >> 5, lane = t & 31;
  const int lhalf = lane >> 4, cn = lane & 15;
  const int QT = SEQ / 16;
  int gid = blockIdx.x * 8 + wave;
  int b = gid / (NH * QT);
  int r = gid % (NH * QT);
  int h = r / QT;
  int q0 = (r % QT) * 16;

  const bf16* Qp = Q + (size_t)(b * NH + h) * SEQ * HD;
  const bf16* Kp = K + (size_t)(b * NH + h) * SEQ * HD;
  const bf16* Vp = Vt + (size_t)(b * NH + h) * HD * SEQ;
  bf16* Pw = &Ps[wave * 16 * 32];

  v16bf qa[4];
#pragma unroll
  for (int f = 0; f < 4; ++f) qa[f] = load_fragA(Qp + (size_t)q0 * HD + f * 32, HD);

  v8f acc[8];
#pragma unroll
  for (int nf = 0; nf < 8; ++nf) acc[nf] = vzero8();
  float mrow[8], lrow[8];
#pragma unroll
  for (int v = 0; v < 8; ++v) { mrow[v] = -3.0e38f; lrow[v] = 0.0f; }

  const float SCALE = 0.08838834764831845f;  // 1/sqrt(128)
  const int nkeys = q0 + 16;
  for (int kb0 = 0; kb0 < nkeys; kb0 += 32) {
    if (kb0 + 32 < SEQ)
      __builtin_prefetch((const void*)(Kp + (size_t)(kb0 + 32) * HD), 0, 1);

    v8f s[2];
#pragma unroll
    for (int kf = 0; kf < 2; ++kf) {
      v8f sv = vzero8();
#pragma unroll
      for (int f = 0; f < 4; ++f) {
        v16bf kbf = load_fragB(Kp + (size_t)(kb0 + kf * 16) * HD + f * 32, HD);
        sv = wmma_bf16(qa[f], kbf, sv);
      }
      s[kf] = sv;
    }

#pragma unroll
    for (int v = 0; v < 8; ++v) {
      const int qrow = q0 + lhalf * 8 + v;
      float x0 = s[0][v] * SCALE, x1 = s[1][v] * SCALE;
      if (kb0 + cn > qrow)      x0 = -1.0e9f;  // causal mask (matches reference -1e9)
      if (kb0 + 16 + cn > qrow) x1 = -1.0e9f;
      float rm = fmaxf(x0, x1);
      rm = fmaxf(rm, __shfl_xor(rm, 1));
      rm = fmaxf(rm, __shfl_xor(rm, 2));
      rm = fmaxf(rm, __shfl_xor(rm, 4));
      rm = fmaxf(rm, __shfl_xor(rm, 8));
      float mnew = fmaxf(mrow[v], rm);
      float sc = __expf(mrow[v] - mnew);
      mrow[v] = mnew;
      float p0 = __expf(x0 - mnew), p1 = __expf(x1 - mnew);
      float rs = p0 + p1;
      rs += __shfl_xor(rs, 1);
      rs += __shfl_xor(rs, 2);
      rs += __shfl_xor(rs, 4);
      rs += __shfl_xor(rs, 8);
      lrow[v] = lrow[v] * sc + rs;
#pragma unroll
      for (int nf = 0; nf < 8; ++nf) acc[nf][v] *= sc;
      Pw[(lhalf * 8 + v) * 32 + cn]      = (bf16)p0;
      Pw[(lhalf * 8 + v) * 32 + 16 + cn] = (bf16)p1;
    }
    // same-wave LDS RAW: ensure P stores land before fragment reload
    asm volatile("s_wait_dscnt 0" ::: "memory");
    v16bf pa = load_fragA(Pw, 32);
#pragma unroll
    for (int nf = 0; nf < 8; ++nf) {
      v16bf vb = load_fragB(Vp + (size_t)(nf * 16) * SEQ + kb0, SEQ);
      acc[nf] = wmma_bf16(pa, vb, acc[nf]);
    }
  }

  float inv[8];
#pragma unroll
  for (int v = 0; v < 8; ++v) inv[v] = 1.0f / lrow[v];
#pragma unroll
  for (int nf = 0; nf < 8; ++nf)
#pragma unroll
    for (int v = 0; v < 8; ++v) {
      int srow = q0 + lhalf * 8 + v;
      O[((size_t)b * SEQ + srow) * DIM + h * HD + nf * 16 + cn] =
          (bf16)(acc[nf][v] * inv[v]);
    }
}

extern "C" void kernel_launch(void* const* d_in, const int* in_sizes, int n_in,
                              void* d_out, int out_size, void* d_ws, size_t ws_size,
                              hipStream_t stream) {
  const float* x  = (const float*)d_in[0];
  const float* Wq = (const float*)d_in[1];
  const float* Wk = (const float*)d_in[2];
  const float* Wv = (const float*)d_in[3];
  const float* Wo = (const float*)d_in[4];
  const float* bo = (const float*)d_in[5];
  float* out = (float*)d_out;

  char* ws = (char*)d_ws;
  size_t off = 0;
  auto carve = [&](size_t bytes) {
    char* p = ws + off;
    off += (bytes + 255) & ~(size_t)255;
    return p;
  };
  bf16* Xb  = (bf16*)carve((size_t)ROWS * DIM * 2);
  bf16* Wqb = (bf16*)carve((size_t)DIM * DIM * 2);
  bf16* Wkb = (bf16*)carve((size_t)DIM * DIM * 2);
  bf16* Wvb = (bf16*)carve((size_t)DIM * DIM * 2);
  bf16* Wob = (bf16*)carve((size_t)DIM * DIM * 2);
  bf16* Qb  = (bf16*)carve((size_t)ROWS * DIM * 2);
  bf16* Kb  = (bf16*)carve((size_t)ROWS * DIM * 2);
  bf16* Vtb = (bf16*)carve((size_t)ROWS * DIM * 2);
  bf16* Ob  = (bf16*)carve((size_t)ROWS * DIM * 2);

  const int xn4 = ROWS * DIM / 4;
  const int wn4 = DIM * DIM / 4;
  cast_f32_bf16<<<xn4 / 256, 256, 0, stream>>>(x, Xb, xn4);
  cast_f32_bf16<<<wn4 / 256, 256, 0, stream>>>(Wq, Wqb, wn4);
  cast_f32_bf16<<<wn4 / 256, 256, 0, stream>>>(Wk, Wkb, wn4);
  cast_f32_bf16<<<wn4 / 256, 256, 0, stream>>>(Wv, Wvb, wn4);
  cast_f32_bf16<<<wn4 / 256, 256, 0, stream>>>(Wo, Wob, wn4);

  dim3 g1(ROWS / 64, DIM / 128, 3);
  gemm_qkvo<<<g1, 256, 0, stream>>>(Xb, Wqb, Wkb, Wvb, Wob, Qb, Kb, Vtb, nullptr, bo, 0);

  flash_attn<<<dim3((ROWS / 16) * NH / 8), 256, 0, stream>>>(Qb, Kb, Vtb, Ob);

  dim3 g2(ROWS / 64, DIM / 128, 1);
  gemm_qkvo<<<g2, 256, 0, stream>>>(Ob, Wqb, Wkb, Wvb, Wob, nullptr, nullptr, nullptr, out, bo, 3);
}
